// PVCNNVelocityNet_88373247082625
// MI455X (gfx1250) — matrix operand order repaired and checked by reading, not compiled, not run on hardware
//
#include <hip/hip_runtime.h>
#include <math.h>

// ---------------- constants ----------------
#define BB 4
#define NPTS 16384
#define EMBC 256

typedef __attribute__((ext_vector_type(16))) _Float16 v16h;
typedef __attribute__((ext_vector_type(8)))  _Float16 h8;
typedef __attribute__((ext_vector_type(8)))  float    v8f;

union AF { v16h v; h8 p[2]; };
union PK2 { _Float16 h[2]; unsigned u; };

// =====================================================================
// WMMA GEMM:  Y[b,o,n] = sum_k W[o,k] * X[b,k,n] + bias[o]
// X (B,K,N), W (O,K), Y (B,O,N).  K MUST be a multiple of 32.
// Block = 128 thr (4 waves), tile = 16 (O) x 64 (N), K chunks of 32.
// LDS: A row-major [16][40] (padded), B col-major [64][40] (padded) so
// each 16x16x32 fragment is two contiguous 16B runs -> ds_load_b128.
// All per-batch offsets fit in 32 bits -> int pointer stepping (no u64
// multiplies in the inner loop).
// =====================================================================
__global__ __launch_bounds__(128) void k_gemm(
    const float* __restrict__ X, const float* __restrict__ W,
    const float* __restrict__ bias, float* __restrict__ Y,
    int K, int O, int N)
{
  __shared__ __align__(16) _Float16 As[16][40];
  __shared__ __align__(16) _Float16 Bs[64][40];   // [n][k] col-major
  int b   = blockIdx.z;
  int o0  = blockIdx.y * 16;
  int n0  = blockIdx.x * 64;
  int tid = threadIdx.x, wave = tid >> 5, lane = tid & 31;
  int half = lane >> 4, l15 = lane & 15;
  int col  = wave*16 + l15;
  const float* Xb = X + (size_t)b * K * N;
  v8f acc = {0.f,0.f,0.f,0.f,0.f,0.f,0.f,0.f};

  int am  = tid >> 4;        // A: row 0..7 (+8 on 2nd it)
  int akp = tid & 15;        // A: k-pair 0..15
  int nb  = tid & 63;        // B: column (constant per thread)
  int kpb = tid >> 6;        // B: k-pair base 0..1

  for (int k0 = 0; k0 < K; k0 += 32) {
    // ---- A tile: 16 x 32, float2 loads, packed b32 stores ----
    {
      const float* pw = W + (o0 + am)*K + k0 + akp*2;   // int offsets
#pragma unroll
      for (int it = 0; it < 2; ++it) {
        float2 v = *(const float2*)pw;
        PK2 pk; pk.h[0] = (_Float16)v.x; pk.h[1] = (_Float16)v.y;
        *(unsigned*)&As[am + it*8][akp*2] = pk.u;
        pw += 8*K;
      }
    }
    // ---- B tile: 32 x 64 -> col-major; pointer-stepped b32 loads ----
    {
      const float* pb = Xb + (k0 + kpb*2)*N + n0 + nb;  // int offsets
#pragma unroll
      for (int it = 0; it < 8; ++it) {
        float v0 = pb[0];
        float v1 = pb[N];
        PK2 pk; pk.h[0] = (_Float16)v0; pk.h[1] = (_Float16)v1;
        *(unsigned*)&Bs[nb][(kpb + it*2)*2] = pk.u;
        pb += 4*N;
      }
      if (k0 + 32 < K) __builtin_prefetch(pb, 0, 0);    // next chunk row
    }
    __syncthreads();
    AF a, bf;
    a.p[0]  = *(const h8*)&As[l15][half*8];
    a.p[1]  = *(const h8*)&As[l15][16 + half*8];
    bf.p[0] = *(const h8*)&Bs[col][half*8];
    bf.p[1] = *(const h8*)&Bs[col][16 + half*8];
    acc = __builtin_amdgcn_wmma_f32_16x16x32_f16(false, a.v, false, bf.v,
                                                 (short)0, acc, false, false);
    __syncthreads();
  }
  int n = n0 + col;
#pragma unroll
  for (int q = 0; q < 8; ++q) {
    int m = o0 + q + 8*half;
    Y[((size_t)b*O + m)*N + n] = acc[q] + bias[m];
  }
}

// =====================================================================
// Weight repack for conv3d:  W (O=C, C, 27)  ->  Wr (27, C, C)
// =====================================================================
__global__ void k_wrepack(const float* __restrict__ W, float* __restrict__ Wr,
                          int C) {
  size_t total = (size_t)C*C*27;
  size_t i = (size_t)blockIdx.x*256 + threadIdx.x;
  if (i >= total) return;
  int cc  = C*C;
  int tap = (int)(i / cc);
  int rem = (int)(i % cc);
  int o = rem / C, c = rem % C;
  Wr[i] = W[((size_t)o*C + c)*27 + tap];
}

// =====================================================================
// WMMA implicit-GEMM 3x3x3 conv + eval BN + leaky(0.1)
// Xv,Yv (B,C,V), Wr (27,C,C) repacked, V = r^3, C multiple of 32
// =====================================================================
__global__ __launch_bounds__(128) void k_conv3d(
    const float* __restrict__ Xv, const float* __restrict__ Wr,
    const float* __restrict__ cb,
    const float* __restrict__ bn_b, const float* __restrict__ bn_rm,
    const float* __restrict__ bn_rv, const float* __restrict__ bn_w,
    float* __restrict__ Yv, int C, int r)
{
  const int V = r*r*r;
  __shared__ __align__(16) _Float16 As[16][40];
  __shared__ __align__(16) _Float16 Bs[64][40];   // [voxel][k]
  __shared__ int ssrc[64];
  __shared__ int sd[64], sh_[64], sw_[64];
  int b   = blockIdx.z;
  int o0  = blockIdx.y * 16;
  int v0  = blockIdx.x * 64;
  int tid = threadIdx.x, wave = tid >> 5, lane = tid & 31;
  int half = lane >> 4, l15 = lane & 15;
  int col  = wave*16 + l15;
  int am  = tid >> 4, akp = tid & 15;
  int jb  = tid & 63, kpb = tid >> 6;
  if (tid < 64) {
    int v = v0 + tid;
    sw_[tid] = v % r; sh_[tid] = (v / r) % r; sd[tid] = v / (r*r);
  }
  const float* Xb = Xv + (size_t)b * C * V;
  v8f acc = {0.f,0.f,0.f,0.f,0.f,0.f,0.f,0.f};

  for (int tap = 0; tap < 27; ++tap) {
    int kd = tap/9, kh = (tap/3)%3, kw = tap%3;
    __syncthreads();
    if (tid < 64) {
      int d2 = sd[tid]+kd-1, h2 = sh_[tid]+kh-1, w2 = sw_[tid]+kw-1;
      ssrc[tid] = (d2>=0 && d2<r && h2>=0 && h2<r && w2>=0 && w2<r)
                    ? ((d2*r+h2)*r+w2) : -1;
    }
    __syncthreads();
    int src = ssrc[jb];                 // constant per thread for this tap
    for (int c0 = 0; c0 < C; c0 += 32) {
      // A tile from repacked weights: contiguous in c -> float2 loads
      {
        const float* pw = Wr + (tap*C + o0 + am)*C + c0 + akp*2;
#pragma unroll
        for (int it = 0; it < 2; ++it) {
          float2 v = *(const float2*)pw;
          PK2 pk; pk.h[0] = (_Float16)v.x; pk.h[1] = (_Float16)v.y;
          *(unsigned*)&As[am + it*8][akp*2] = pk.u;
          pw += 8*C;
        }
      }
      // B tile: gathered voxel columns; boundary branch hoisted
      if (src >= 0) {
        const float* pbv = Xb + (c0 + kpb*2)*V + src;
#pragma unroll
        for (int it = 0; it < 8; ++it) {
          float v0f = pbv[0];
          float v1f = pbv[V];
          PK2 pk; pk.h[0] = (_Float16)v0f; pk.h[1] = (_Float16)v1f;
          *(unsigned*)&Bs[jb][(kpb + it*2)*2] = pk.u;
          pbv += 4*V;
        }
      } else {
#pragma unroll
        for (int it = 0; it < 8; ++it)
          *(unsigned*)&Bs[jb][(kpb + it*2)*2] = 0u;
      }
      __syncthreads();
      AF a, bf;
      a.p[0]  = *(const h8*)&As[l15][half*8];
      a.p[1]  = *(const h8*)&As[l15][16 + half*8];
      bf.p[0] = *(const h8*)&Bs[col][half*8];
      bf.p[1] = *(const h8*)&Bs[col][16 + half*8];
      acc = __builtin_amdgcn_wmma_f32_16x16x32_f16(false, a.v, false, bf.v,
                                                   (short)0, acc, false, false);
      __syncthreads();
    }
  }
  int vcol = v0 + col;
#pragma unroll
  for (int q = 0; q < 8; ++q) {
    int m = o0 + q + 8*half;
    float val = acc[q] + cb[m];
    float sc  = bn_w[m] * rsqrtf(bn_rv[m] + 1e-4f);
    val = (val - bn_rm[m]) * sc + bn_b[m];
    val = (val > 0.f) ? val : 0.1f * val;
    Yv[((size_t)b*C + m)*V + vcol] = val;
  }
}

// ---------------- small / glue kernels ----------------

__global__ void k_embed(const float* t, const float* cond,
                        const float* tw, const float* tb,
                        const float* cw, const float* cb, float* emb)
{
  __shared__ float traw[BB][EMBC];
  int tid = threadIdx.x;
  for (int b = 0; b < BB; ++b) {
    if (tid < 128) {
      float fr = expf(-logf(10000.f) * (float)tid / 128.f);
      float a  = t[b] * fr;
      traw[b][tid]       = cosf(a);
      traw[b][tid + 128] = sinf(a);
    }
  }
  __syncthreads();
  int o = tid;
  for (int b = 0; b < BB; ++b) {
    float s = tb[o];
    for (int k = 0; k < EMBC; ++k) s += traw[b][k] * tw[o*EMBC + k];
    float te = s / (1.f + expf(-s));
    float s2 = cb[o];
    for (int k = 0; k < 512; ++k) s2 += cond[b*512 + k] * cw[o*512 + k];
    float ce = s2 / (1.f + expf(-s2));
    emb[b*EMBC + o] = te + ce;
  }
}

__global__ void k_coords(const float* x, float* coords, int N) {
  int i = blockIdx.x*256 + threadIdx.x;
  if (i >= BB*N) return;
  int b = i / N, n = i % N;
  for (int d = 0; d < 3; ++d)
    coords[((size_t)b*3 + d)*N + n] = x[((size_t)b*N + n)*3 + d];
}

__global__ void k_s0_bias(const float* emb, const float* W, const float* pb,
                          float* s0b) {
  int t = threadIdx.x;              // 256 == B*64
  int b = t >> 6, o = t & 63;
  float s = pb[o];
  for (int k = 0; k < EMBC; ++k) s += emb[b*EMBC + k] * W[o*259 + k];
  s0b[t] = s;
}

__global__ void k_s0_apply(const float* coords, const float* s0b,
                           const float* W, float* out, int N) {
  int i = blockIdx.x*256 + threadIdx.x;
  if (i >= BB*N) return;
  int b = i / N, n = i % N;
  float cx = coords[((size_t)b*3+0)*N + n];
  float cy = coords[((size_t)b*3+1)*N + n];
  float cz = coords[((size_t)b*3+2)*N + n];
  for (int o = 0; o < 64; ++o) {
    const float* w = W + o*259 + 256;
    out[((size_t)b*64 + o)*N + n] = s0b[b*64+o] + w[0]*cx + w[1]*cy + w[2]*cz;
  }
}

__global__ void k_gn_stats(const float* Y, float* stats, int C, int N) {
  __shared__ float s1[256], s2[256];
  int bg = blockIdx.x, b = bg >> 5, g = bg & 31;
  int cpg = C / 32;
  size_t base = ((size_t)b*C + (size_t)g*cpg) * N;
  size_t total = (size_t)cpg * N;
  float a = 0.f, q = 0.f;
  for (size_t i = threadIdx.x; i < total; i += 256) {
    float v = Y[base + i]; a += v; q += v*v;
  }
  s1[threadIdx.x] = a; s2[threadIdx.x] = q; __syncthreads();
  for (int st = 128; st > 0; st >>= 1) {
    if (threadIdx.x < st) { s1[threadIdx.x] += s1[threadIdx.x+st];
                            s2[threadIdx.x] += s2[threadIdx.x+st]; }
    __syncthreads();
  }
  if (threadIdx.x == 0) {
    float m = s1[0] / (float)total;
    float var = s2[0] / (float)total - m*m;
    stats[bg*2]   = m;
    stats[bg*2+1] = rsqrtf(var + 1e-5f);
  }
}

// act: 0 none, 1 relu, 2 silu  (in-place)
__global__ void k_gn_act(float* y, const float* stats, const float* w,
                         const float* b_, int C, int N, int act) {
  size_t i = (size_t)blockIdx.x*256 + threadIdx.x;
  if (i >= (size_t)BB*C*N) return;
  size_t t = i / N; int c = (int)(t % C); int b = (int)(t / C);
  int g = c / (C/32);
  float m = stats[(b*32+g)*2], inv = stats[(b*32+g)*2+1];
  float v = (y[i]-m)*inv*w[c] + b_[c];
  if (act == 1) v = fmaxf(v, 0.f);
  else if (act == 2) v = v / (1.f + expf(-v));
  y[i] = v;
}

__global__ void k_gn_film(const float* f, const float* stats, const float* gnw,
                          const float* gnb, const float* gb, float* out,
                          int C, int N) {
  size_t i = (size_t)blockIdx.x*256 + threadIdx.x;
  if (i >= (size_t)BB*C*N) return;
  size_t t = i / N; int c = (int)(t % C); int b = (int)(t / C);
  int g = c / (C/32);
  float m = stats[(b*32+g)*2], inv = stats[(b*32+g)*2+1];
  float v = f[i];
  float norm = (v - m)*inv*gnw[c] + gnb[c];
  out[i] = v + norm*gb[b*2*C + c] + gb[b*2*C + C + c];
}

__global__ void k_film_aff(const float* emb, const float* aw, const float* ab,
                           float* gb, int C2) {
  int e = blockIdx.x*256 + threadIdx.x;
  if (e >= BB*C2) return;
  int b = e / C2, j = e % C2;
  float s = ab[j];
  for (int k = 0; k < EMBC; ++k) s += emb[b*EMBC + k] * aw[j*EMBC + k];
  gb[b*C2 + j] = s;
}

__global__ void k_cmean(const float* coords, float* cm, int N) {
  __shared__ float sh[256];
  int bd = blockIdx.x;
  const float* p = coords + (size_t)bd * N;
  float s = 0.f;
  for (int i = threadIdx.x; i < N; i += 256) s += p[i];
  sh[threadIdx.x] = s; __syncthreads();
  for (int st = 128; st > 0; st >>= 1) {
    if (threadIdx.x < st) sh[threadIdx.x] += sh[threadIdx.x+st];
    __syncthreads();
  }
  if (threadIdx.x == 0) cm[bd] = sh[0] / (float)N;
}

__global__ void k_cmax(const float* coords, const float* cm, float* cinv, int N) {
  __shared__ float sh[256];
  int b = blockIdx.x;
  float mx = 0.f;
  for (int n = threadIdx.x; n < N; n += 256) {
    float dx = coords[((size_t)b*3+0)*N+n] - cm[b*3+0];
    float dy = coords[((size_t)b*3+1)*N+n] - cm[b*3+1];
    float dz = coords[((size_t)b*3+2)*N+n] - cm[b*3+2];
    float nr = sqrtf(dx*dx + dy*dy + dz*dz);
    mx = fmaxf(mx, nr);
  }
  sh[threadIdx.x] = mx; __syncthreads();
  for (int st = 128; st > 0; st >>= 1) {
    if (threadIdx.x < st) sh[threadIdx.x] = fmaxf(sh[threadIdx.x], sh[threadIdx.x+st]);
    __syncthreads();
  }
  if (threadIdx.x == 0) cinv[b] = 1.f / (2.f*sh[0] + 1e-6f);
}

__global__ void k_voxidx(const float* coords, const float* cm, const float* cinv,
                         float* nc, int* idx, int r, int N) {
  int i = blockIdx.x*256 + threadIdx.x;
  if (i >= BB*N) return;
  int b = i / N, n = i % N;
  int vc[3];
  for (int d = 0; d < 3; ++d) {
    float c = (coords[((size_t)b*3+d)*N+n] - cm[b*3+d]) * cinv[b] + 0.5f;
    float v = fminf(fmaxf(c * (float)r, 0.f), (float)(r-1));
    nc[((size_t)b*3+d)*N+n] = v;
    vc[d] = (int)rintf(v);
  }
  idx[b*N + n] = (vc[0]*r + vc[1])*r + vc[2];
}

__global__ void k_zero(float* p, size_t total) {
  size_t i = (size_t)blockIdx.x*256 + threadIdx.x;
  if (i < total) p[i] = 0.f;
}

__global__ void k_scatter(const float* f, const int* idx, float* vox,
                          float* cnt, int C, int V, int N) {
  int i = blockIdx.x*256 + threadIdx.x;
  if (i >= BB*N) return;
  int b = i / N, n = i % N;
  int id = idx[b*N + n];
  atomicAdd(&cnt[(size_t)b*V + id], 1.f);
  const float* fb = f + (size_t)b*C*N + n;
  float* vb = vox + (size_t)b*C*V + id;
  for (int c = 0; c < C; ++c)
    atomicAdd(vb + (size_t)c*V, fb[(size_t)c*N]);
}

__global__ void k_voxdiv(float* vox, const float* cnt, int C, int V) {
  size_t i = (size_t)blockIdx.x*256 + threadIdx.x;
  if (i >= (size_t)BB*C*V) return;
  int v = (int)(i % V); int b = (int)(i / ((size_t)C*V));
  vox[i] /= fmaxf(cnt[(size_t)b*V + v], 1.f);
}

__global__ void k_se_mean(const float* vox, float* mean, int V) {
  __shared__ float sh[256];
  int bc = blockIdx.x;
  const float* p = vox + (size_t)bc * V;
  float s = 0.f;
  for (int i = threadIdx.x; i < V; i += 256) s += p[i];
  sh[threadIdx.x] = s; __syncthreads();
  for (int st = 128; st > 0; st >>= 1) {
    if (threadIdx.x < st) sh[threadIdx.x] += sh[threadIdx.x+st];
    __syncthreads();
  }
  if (threadIdx.x == 0) mean[bc] = sh[0] / (float)V;
}

__global__ void k_se_mlp(const float* mean, const float* w1, const float* b1,
                         const float* w2, const float* b2, float* g, int C) {
  int Ch = C / 8;
  __shared__ float hid[BB*16];
  int t = threadIdx.x;
  if (t < BB*Ch) {
    int b = t / Ch, h = t % Ch;
    float s = b1[h];
    for (int c = 0; c < C; ++c) s += mean[b*C + c] * w1[h*C + c];
    hid[b*Ch + h] = fmaxf(s, 0.f);
  }
  __syncthreads();
  for (int e = t; e < BB*C; e += 256) {
    int b = e / C, c = e % C;
    float s = b2[c];
    for (int h = 0; h < Ch; ++h) s += hid[b*Ch + h] * w2[c*Ch + h];
    g[e] = 1.f / (1.f + expf(-s));
  }
}

__global__ void k_se_scale(float* vox, const float* g, int C, int V) {
  size_t i = (size_t)blockIdx.x*256 + threadIdx.x;
  if (i >= (size_t)BB*C*V) return;
  size_t bc = i / V;
  vox[i] *= g[bc];
}

__global__ void k_devox_add(const float* vox, const float* nc, float* out,
                            int C, int r, int N) {
  int i = blockIdx.x*256 + threadIdx.x;
  if (i >= BB*N) return;
  int b = i / N, n = i % N;
  int V = r*r*r;
  float px = nc[((size_t)b*3+0)*N+n];
  float py = nc[((size_t)b*3+1)*N+n];
  float pz = nc[((size_t)b*3+2)*N+n];
  int lx = (int)floorf(px), ly = (int)floorf(py), lz = (int)floorf(pz);
  int hx = min(lx+1, r-1), hy = min(ly+1, r-1), hz = min(lz+1, r-1);
  float wx = px - (float)lx, wy = py - (float)ly, wz = pz - (float)lz;
  int   fl[8]; float wt[8]; int q = 0;
  for (int dx = 0; dx < 2; ++dx)
    for (int dy = 0; dy < 2; ++dy)
      for (int dz = 0; dz < 2; ++dz) {
        int xi = dx ? hx : lx, yi = dy ? hy : ly, zi = dz ? hz : lz;
        float w = (dx ? wx : 1.f-wx) * (dy ? wy : 1.f-wy) * (dz ? wz : 1.f-wz);
        fl[q] = (xi*r + yi)*r + zi; wt[q] = w; ++q;
      }
  const float* vb = vox + (size_t)b*C*V;
  float* ob = out + (size_t)b*C*N;
  for (int c = 0; c < C; ++c) {
    const float* vc = vb + (size_t)c*V;
    float s = 0.f;
    for (int k = 0; k < 8; ++k) s += vc[fl[k]] * wt[k];
    ob[(size_t)c*N + n] += s;
  }
}

__global__ void k_gmax(const float* f, float* gmax, int N) {
  __shared__ float sh[256];
  int bc = blockIdx.x;
  const float* p = f + (size_t)bc * N;
  float mx = -3.4e38f;
  for (int i = threadIdx.x; i < N; i += 256) mx = fmaxf(mx, p[i]);
  sh[threadIdx.x] = mx; __syncthreads();
  for (int st = 128; st > 0; st >>= 1) {
    if (threadIdx.x < st) sh[threadIdx.x] = fmaxf(sh[threadIdx.x], sh[threadIdx.x+st]);
    __syncthreads();
  }
  if (threadIdx.x == 0) gmax[bc] = sh[0];
}

__global__ void k_gmlp(const float* gmax, const float* w1, const float* b1,
                       const float* w2, const float* b2, float* gv) {
  __shared__ float hid[BB*128];
  int t = threadIdx.x;
  for (int e = t; e < BB*128; e += 256) {
    int b = e >> 7, h = e & 127;
    float s = b1[h];
    for (int c = 0; c < 128; ++c) s += gmax[b*128 + c] * w1[h*128 + c];
    hid[e] = s / (1.f + expf(-s));
  }
  __syncthreads();
  for (int e = t; e < BB*128; e += 256) {
    int b = e >> 7, c = e & 127;
    float s = b2[c];
    for (int h = 0; h < 128; ++h) s += hid[b*128 + h] * w2[c*128 + h];
    gv[e] = s;
  }
}

__global__ void k_concat(const float* f, const float* gv, float* fc, int N) {
  size_t i = (size_t)blockIdx.x*256 + threadIdx.x;
  if (i >= (size_t)BB*256*N) return;
  int n = (int)(i % N); size_t t = i / N;
  int c = (int)(t % 256); int b = (int)(t / 256);
  fc[i] = (c < 128) ? f[((size_t)b*128 + c)*N + n] : gv[b*128 + (c - 128)];
}

__global__ void k_head_out(const float* h, const float* W, const float* b_,
                           float* out, int N) {
  int i = blockIdx.x*256 + threadIdx.x;
  if (i >= BB*N) return;
  int b = i / N, n = i % N;
  const float* hb = h + (size_t)b*128*N + n;
  for (int o = 0; o < 3; ++o) {
    float s = b_[o];
    for (int c = 0; c < 128; ++c) s += hb[(size_t)c*N] * W[o*128 + c];
    out[((size_t)b*N + n)*3 + o] = s;
  }
}

// ---------------- host orchestration ----------------

struct StageP {
  const float *film_aff_b, *film_aff_w, *film_gn_b, *film_gn_w;
  const float *post_b, *post_gn_b, *post_gn_w, *post_w;
  const float *bn1_b, *bn1_rm, *bn1_rv, *bn1_w;
  const float *bn2_b, *bn2_rm, *bn2_rv, *bn2_w;
  const float *conv1_b, *conv1_w, *conv2_b, *conv2_w;
  const float *point_b, *point_gn_b, *point_gn_w, *point_w;
  const float *se_b1, *se_b2, *se_w1, *se_w2;
  const float *proj_b, *proj_gn_b, *proj_gn_w, *proj_w;
};

extern "C" void kernel_launch(void* const* d_in, const int* in_sizes, int n_in,
                              void* d_out, int out_size, void* d_ws, size_t ws_size,
                              hipStream_t stream) {
  const int N = NPTS;
  const float* x    = (const float*)d_in[0];
  const float* t    = (const float*)d_in[1];
  const float* cond = (const float*)d_in[2];
  int pi = 3;
  auto nx = [&]() { return (const float*)d_in[pi++]; };
  // jax tree-flatten order: dicts by sorted key, lists in order.
  const float* cproj_b = nx(); const float* cproj_w = nx();
  const float* glb_b1  = nx(); const float* glb_b2  = nx();
  const float* glb_w1  = nx(); const float* glb_w2  = nx();
  const float* hgn_b   = nx(); const float* hgn_w   = nx();
  const float* hout_b  = nx(); const float* hout_w  = nx();
  const float* hpre_b  = nx(); const float* hpre_w  = nx();
  StageP st[3];
  for (int s = 0; s < 3; ++s) {
    StageP& q = st[s];
    q.film_aff_b = nx(); q.film_aff_w = nx(); q.film_gn_b = nx(); q.film_gn_w = nx();
    q.post_b = nx(); q.post_gn_b = nx(); q.post_gn_w = nx(); q.post_w = nx();
    q.bn1_b = nx(); q.bn1_rm = nx(); q.bn1_rv = nx(); q.bn1_w = nx();
    q.bn2_b = nx(); q.bn2_rm = nx(); q.bn2_rv = nx(); q.bn2_w = nx();
    q.conv1_b = nx(); q.conv1_w = nx(); q.conv2_b = nx(); q.conv2_w = nx();
    q.point_b = nx(); q.point_gn_b = nx(); q.point_gn_w = nx(); q.point_w = nx();
    q.se_b1 = nx(); q.se_b2 = nx(); q.se_w1 = nx(); q.se_w2 = nx();
    q.proj_b = nx(); q.proj_gn_b = nx(); q.proj_gn_w = nx(); q.proj_w = nx();
  }
  const float* tproj_b = nx(); const float* tproj_w = nx();

  // -------- workspace arena (floats) --------
  float* ws = (float*)d_ws;
  const size_t BIG = 8388608;            // 4*128*16384 and 4*64*32768
  float* f_a    = ws;
  float* f_b    = f_a + BIG;
  float* vox_a  = f_b + BIG;
  float* vox_b  = vox_a + BIG;
  float* fcat   = vox_a;                 // alias: 16.7M floats over vox_a+vox_b
  float* cnt    = vox_b + BIG;           // 4*32768
  float* ncbuf  = cnt + 131072;          // 4*3*N
  int*   idx    = (int*)(ncbuf + 196608);
  float* coords = (float*)idx + 65536;   // 4*3*N
  float* emb    = coords + 196608;       // 4*256
  float* stats  = emb + 1024;            // 4*32*2
  float* cm     = stats + 256;           // 12
  float* cinv   = cm + 16;               // 4
  float* s0bias = cinv + 16;             // 256
  float* semean = s0bias + 256;          // 512
  float* seg    = semean + 512;          // 512
  float* gbbuf  = seg + 512;             // 4*256
  float* gmax   = gbbuf + 1024;          // 512
  float* gv     = gmax + 512;            // 512
  float* wrep   = gv + 512;              // 27*128*128 = 442368 (repacked conv w)

  auto run_gn = [&](float* buf, const float* w, const float* b_, int C, int act) {
    k_gn_stats<<<BB*32, 256, 0, stream>>>(buf, stats, C, N);
    size_t tot = (size_t)BB*C*N;
    k_gn_act<<<(unsigned)((tot+255)/256), 256, 0, stream>>>(buf, stats, w, b_, C, N, act);
  };

  // -------- embed + coords --------
  k_embed<<<1, 256, 0, stream>>>(t, cond, tproj_w, tproj_b, cproj_w, cproj_b, emb);
  k_coords<<<(BB*N+255)/256, 256, 0, stream>>>(x, coords, N);

  // -------- stage 0 projection (259 -> 64, emb folded into bias) --------
  k_s0_bias<<<1, 256, 0, stream>>>(emb, st[0].proj_w, st[0].proj_b, s0bias);
  k_s0_apply<<<(BB*N+255)/256, 256, 0, stream>>>(coords, s0bias, st[0].proj_w, f_a, N);
  run_gn(f_a, st[0].proj_gn_w, st[0].proj_gn_b, 64, 1);

  const int Cs[3] = {64, 128, 128};
  const int Rs[3] = {32, 16, 8};
  for (int s = 0; s < 3; ++s) {
    int C = Cs[s], r = Rs[s], V = r*r*r;
    if (s > 0) {                          // stage projection GEMM f_b -> f_a
      dim3 g(N/64, C/16, BB);
      k_gemm<<<g, 128, 0, stream>>>(f_b, st[s].proj_w, st[s].proj_b, f_a,
                                    Cs[s-1], C, N);
      run_gn(f_a, st[s].proj_gn_w, st[s].proj_gn_b, C, 1);
    }
    // ---- voxelize f_a ----
    k_cmean<<<BB*3, 256, 0, stream>>>(coords, cm, N);
    k_cmax <<<BB,   256, 0, stream>>>(coords, cm, cinv, N);
    k_voxidx<<<(BB*N+255)/256, 256, 0, stream>>>(coords, cm, cinv, ncbuf, idx, r, N);
    size_t vt = (size_t)BB*C*V;
    k_zero<<<(unsigned)((vt+255)/256), 256, 0, stream>>>(vox_a, vt);
    k_zero<<<(BB*V+255)/256, 256, 0, stream>>>(cnt, (size_t)BB*V);
    k_scatter<<<(BB*N+255)/256, 256, 0, stream>>>(f_a, idx, vox_a, cnt, C, V, N);
    k_voxdiv<<<(unsigned)((vt+255)/256), 256, 0, stream>>>(vox_a, cnt, C, V);
    // ---- conv3d x2 (WMMA implicit GEMM, repacked weights) ----
    size_t wt_ = (size_t)C*C*27;
    dim3 gc(V/64, C/16, BB);
    k_wrepack<<<(unsigned)((wt_+255)/256), 256, 0, stream>>>(st[s].conv1_w, wrep, C);
    k_conv3d<<<gc, 128, 0, stream>>>(vox_a, wrep, st[s].conv1_b,
        st[s].bn1_b, st[s].bn1_rm, st[s].bn1_rv, st[s].bn1_w, vox_b, C, r);
    k_wrepack<<<(unsigned)((wt_+255)/256), 256, 0, stream>>>(st[s].conv2_w, wrep, C);
    k_conv3d<<<gc, 128, 0, stream>>>(vox_b, wrep, st[s].conv2_b,
        st[s].bn2_b, st[s].bn2_rm, st[s].bn2_rv, st[s].bn2_w, vox_a, C, r);
    // ---- SE ----
    k_se_mean<<<BB*C, 256, 0, stream>>>(vox_a, semean, V);
    k_se_mlp<<<1, 256, 0, stream>>>(semean, st[s].se_w1, st[s].se_b1,
                                    st[s].se_w2, st[s].se_b2, seg, C);
    k_se_scale<<<(unsigned)((vt+255)/256), 256, 0, stream>>>(vox_a, seg, C, V);
    // ---- point branch (WMMA GEMM) then devox add ----
    dim3 gp(N/64, C/16, BB);
    k_gemm<<<gp, 128, 0, stream>>>(f_a, st[s].point_w, st[s].point_b, f_b, C, C, N);
    run_gn(f_b, st[s].point_gn_w, st[s].point_gn_b, C, 1);
    k_devox_add<<<(BB*N+255)/256, 256, 0, stream>>>(vox_a, ncbuf, f_b, C, r, N);
    // ---- post MLP f_b -> f_a ----
    k_gemm<<<gp, 128, 0, stream>>>(f_b, st[s].post_w, st[s].post_b, f_a, C, C, N);
    run_gn(f_a, st[s].post_gn_w, st[s].post_gn_b, C, 1);
    // ---- FiLM: f_b = f_a + gn(f_a)*gamma + beta ----
    k_film_aff<<<(BB*2*C+255)/256, 256, 0, stream>>>(emb, st[s].film_aff_w,
                                                     st[s].film_aff_b, gbbuf, 2*C);
    k_gn_stats<<<BB*32, 256, 0, stream>>>(f_a, stats, C, N);
    size_t tot = (size_t)BB*C*N;
    k_gn_film<<<(unsigned)((tot+255)/256), 256, 0, stream>>>(f_a, stats,
        st[s].film_gn_w, st[s].film_gn_b, gbbuf, f_b, C, N);
  }

  // -------- global branch + head --------
  k_gmax<<<BB*128, 256, 0, stream>>>(f_b, gmax, N);
  k_gmlp<<<1, 256, 0, stream>>>(gmax, glb_w1, glb_b1, glb_w2, glb_b2, gv);
  size_t ct = (size_t)BB*256*N;
  k_concat<<<(unsigned)((ct+255)/256), 256, 0, stream>>>(f_b, gv, fcat, N);
  dim3 gh(N/64, 128/16, BB);
  k_gemm<<<gh, 128, 0, stream>>>(fcat, hpre_w, hpre_b, f_a, 256, 128, N);
  run_gn(f_a, hgn_w, hgn_b, 128, 2);                    // gn + silu
  k_head_out<<<(BB*N+255)/256, 256, 0, stream>>>(f_a, hout_w, hout_b,
                                                 (float*)d_out, N);
}
